// LSA_16114717295133
// MI455X (gfx1250) — matrix operand order repaired and checked by reading, not compiled, not run on hardware
//
#include <hip/hip_runtime.h>
#include <hip/hip_bf16.h>

// ---------------------------------------------------------------------------
// Fused Swin window attention for gfx1250 (MI455X, wave32, WMMA f16->f32)
//   pre-pass: W fp32 -> f16 copy in d_ws (enables TDM byte-copy staging)
//   main:     qkv = window(x) @ W^T + b   (WMMA 16x16x32 f16, f32 accum)
//             W tiles staged via TENSOR_LOAD_TO_LDS (TDM, pad-strided rows)
//             S = q@k^T -> register softmax -> O = P@v -> window-reverse
// One workgroup per window (2048 blocks), 8 wave32 per block.
// ---------------------------------------------------------------------------

typedef __attribute__((ext_vector_type(16))) _Float16     v16h;
typedef __attribute__((ext_vector_type(8)))  float        v8f;
typedef __attribute__((ext_vector_type(4)))  unsigned int v4u;
typedef __attribute__((ext_vector_type(8)))  int          v8i;
typedef __attribute__((ext_vector_type(4)))  int          v4i;

#define CH     256
#define N3     768
#define HC     8
#define NTOK   49     // 7*7 tokens per window
#define HHW    56
#define STRQ   776    // sQKV row stride in halfs (768 + 8 pad)
#define STRXF  264    // resident X tile stride in halfs (256 + 8 pad)
#define STRW   40     // W staging stride in halfs (32 + 8 pad via TDM padding)
#define STRP   68     // P scratch stride (64 + 4 pad)

#define SQKV_BYTES (64 * STRQ * 2)                 // 99,328
#define SXF_BYTES  (64 * STRXF * 2)                // 33,792
#define SW_BYTES   (256 * STRW * 2)                // 20,480
#define SP_BYTES   (8 * 64 * STRP * 2)             // 69,632
#define SMEM_BYTES (SQKV_BYTES + SP_BYTES)         // 168,960 (< 320KB/WG)

// A-fragment 16x32 f16 (ISA 7.12.2): lane 0-15 -> M=lane, K pairs
// {0..7,16..23}; lane 16-31 -> M=lane-16, K pairs {8..15,24..31}.
__device__ __forceinline__ v16h load_frag_a(const _Float16* base, int strM,
                                            int m0, int k0, int kvalid, int lane) {
  const int m    = m0 + (lane & 15);
  const int koff = (lane >> 4) << 3;          // 0 or 8
  const _Float16* row = base + m * strM + k0;
  v16h a;
#pragma unroll
  for (int r = 0; r < 8; ++r) {
    const int kb = ((r >> 2) << 4) + koff + ((r & 3) << 1);
    a[2 * r]     = (kb     < kvalid) ? row[kb]     : (_Float16)0.0f;
    a[2 * r + 1] = (kb + 1 < kvalid) ? row[kb + 1] : (_Float16)0.0f;
  }
  return a;
}

// B-fragment 32x16 f16 (KxN): lane = N (mod 16), K pairs {0..7,16..23} for
// lanes 0-15, {8..15,24..31} for lanes 16-31.  elem(k,n) = base[(n0+n)*strN + (k0+k)*strK].
__device__ __forceinline__ v16h load_frag_b(const _Float16* base, int strN, int strK,
                                            int n0, int k0, int kvalid, int lane) {
  const int n     = n0 + (lane & 15);
  const int khalf = (lane >> 4) << 3;         // 0 or 8
  const _Float16* col = base + n * strN;
  v16h b;
#pragma unroll
  for (int r = 0; r < 8; ++r) {
    const int k = ((r >> 2) << 4) + khalf + ((r & 3) << 1);
    b[2 * r]     = (k     < kvalid) ? col[(k0 + k)     * strK] : (_Float16)0.0f;
    b[2 * r + 1] = (k + 1 < kvalid) ? col[(k0 + k + 1) * strK] : (_Float16)0.0f;
  }
  return b;
}

__device__ __forceinline__ float wave_half_max(float v) {
#pragma unroll
  for (int m = 1; m <= 8; m <<= 1) v = fmaxf(v, __shfl_xor(v, m, 32));
  return v;
}
__device__ __forceinline__ float wave_half_sum(float v) {
#pragma unroll
  for (int m = 1; m <= 8; m <<= 1) v += __shfl_xor(v, m, 32);
  return v;
}

// --------- pre-pass: W fp32 -> f16 (eight elements per thread) -------------
__global__ __launch_bounds__(256) void convert_w_f16(const float* __restrict__ w,
                                                     _Float16* __restrict__ dst) {
  const int i = (blockIdx.x * 256 + threadIdx.x) * 8;   // 196,608 total
#pragma unroll
  for (int j = 0; j < 8; ++j) dst[i + j] = (_Float16)w[i + j];
}

__global__ __launch_bounds__(256) void swin_window_attn_gfx1250(
    const float* __restrict__ x, const _Float16* __restrict__ wsW,
    const float* __restrict__ bqkv, float* __restrict__ out) {
  __shared__ __align__(16) unsigned char smem[SMEM_BYTES];
  _Float16* sQKV = (_Float16*)smem;
  _Float16* sXf  = (_Float16*)(smem + SQKV_BYTES);               // phase 1 (resident X)
  _Float16* sW   = (_Float16*)(smem + SQKV_BYTES + SXF_BYTES);   // phase 1 (TDM target)
  _Float16* sP   = (_Float16*)(smem + SQKV_BYTES);               // phase 2 (aliases)

  const int tid  = threadIdx.x;
  const int lane = tid & 31;
  const int wave = tid >> 5;
  const int win  = blockIdx.x;          // 0..2047
  const int bb   = win >> 6;            // batch
  const int p    = win & 63;            // window in batch
  const int hpi  = p >> 3, wpi = p & 7;
  const long long xbase = (long long)bb * (HHW * HHW) * CH;

  // ---- stage the whole 64x256 window slice of x once (f32 -> f16) ---------
  {
    const int row = tid >> 2;            // 0..63
    const int cg  = (tid & 3) * 64;      // 64 channels per thread
    _Float16* dst = sXf + row * STRXF + cg;
    if (row < NTOK) {
      const int wh = row / 7, ww = row % 7;
      const int tok = (hpi * 7 + wh) * HHW + (wpi * 7 + ww);
      const float* src = x + xbase + (long long)tok * CH + cg;
#pragma unroll
      for (int j = 0; j < 64; ++j) dst[j] = (_Float16)src[j];
    } else {
#pragma unroll
      for (int j = 0; j < 64; ++j) dst[j] = (_Float16)0.0f;
    }
  }
  __syncthreads();

  // ---------------- Phase 1: fused QKV projection (M=64, N=768, K=256) -----
  const float qscale = 0.35355339059327379f;   // hc^-0.5
  const unsigned sW_lds = (unsigned)(unsigned long long)(uintptr_t)sW;

#pragma unroll 1
  for (int nchunk = 0; nchunk < 3; ++nchunk) {
    v8f acc[4][2];
#pragma unroll
    for (int mt = 0; mt < 4; ++mt)
#pragma unroll
      for (int nt = 0; nt < 2; ++nt) acc[mt][nt] = (v8f){};

#pragma unroll 1
    for (int ks = 0; ks < 8; ++ks) {
      const int k0 = ks * 32;

      // ---- TDM: DMA W tile (256 rows x 32 halfs, row pad 4 dwords) --------
      if (wave == 0) {
        const unsigned long long ga =
            (unsigned long long)(uintptr_t)wsW +
            ((unsigned long long)(nchunk * 256) * CH + (unsigned)k0) * 2ull;
        v4u g0;
        g0.x = 1u;                                   // count=1, user mode
        g0.y = sW_lds;                               // LDS byte address
        g0.z = (unsigned)ga;                         // global_addr[31:0]
        g0.w = (unsigned)((ga >> 32) & 0x1FFFFFFull) // global_addr[56:32]
               | (2u << 30);                         // type=2 ("image")
        v8i g1;
        g1[0] = (int)((1u << 16)        // data_size = 2 bytes
                    | (1u << 20)        // pad_enable
                    | (3u << 22)        // pad_interval: 16 DWORDs (=1 row, 64B)
                    | (3u << 25));      // pad_amount:   4 DWORDs (=16B)
        g1[1] = (int)(256u << 16);      // tensor_dim0 = 256 (bits[79:48] lo16)
        g1[2] = (int)(768u << 16);      // dim0 hi16=0 | tensor_dim1=768 lo16
        g1[3] = (int)(32u << 16);       // dim1 hi16=0 | tile_dim0 = 32
        g1[4] = (int)256u;              // tile_dim1 = 256, tile_dim2 = 0
        g1[5] = (int)256u;              // tensor_dim0_stride = 256 (lo32)
        g1[6] = 0;                      // stride hi16 | dim1_stride lo16
        g1[7] = 0;
        const v4i gz4 = (v4i){};
        const v8i gz8 = (v8i){};
        __builtin_amdgcn_tensor_load_to_lds(g0, g1, gz4, gz4, gz8, 0);
        __builtin_amdgcn_s_wait_tensorcnt(0);
      }
      __syncthreads();

      v16h af[4];
#pragma unroll
      for (int mt = 0; mt < 4; ++mt)
        af[mt] = load_frag_a(sXf, STRXF, mt * 16, k0, 32, lane);
#pragma unroll
      for (int nt = 0; nt < 2; ++nt) {
        const v16h bf = load_frag_b(sW, STRW, 1, (wave * 2 + nt) * 16, 0, 32, lane);
#pragma unroll
        for (int mt = 0; mt < 4; ++mt)
          acc[mt][nt] = __builtin_amdgcn_wmma_f32_16x16x32_f16(
              false, af[mt], false, bf, (short)0, acc[mt][nt], false, false);
      }
      __syncthreads();
    }

    // bias add (+ q pre-scale), convert to f16 into LDS-resident QKV
#pragma unroll
    for (int mt = 0; mt < 4; ++mt)
#pragma unroll
      for (int nt = 0; nt < 2; ++nt) {
        const int ncol = nchunk * 256 + (wave * 2 + nt) * 16 + (lane & 15);
        const float bias = bqkv[ncol];
        const float mul  = (ncol < CH) ? qscale : 1.0f;
#pragma unroll
        for (int e = 0; e < 8; ++e) {
          const int m = mt * 16 + e + ((lane >> 4) << 3);
          sQKV[m * STRQ + ncol] = (_Float16)((acc[mt][nt][e] + bias) * mul);
        }
      }
  }
  __syncthreads();

  // ---------------- Phase 2: attention, 4 heads per wave --------------------
  _Float16* myP = sP + wave * (64 * STRP);
#pragma unroll 1
  for (int hh = 0; hh < 4; ++hh) {
    const int head = wave * 4 + hh;
    const _Float16* qb = sQKV + head * HC;            // q[i][d]
    const _Float16* kb = sQKV + CH + head * HC;       // k[m][d]
    const _Float16* vb = sQKV + 2 * CH + head * HC;   // v[m][d]

    // S = q @ k^T  (K = 8 padded to 32 via zero-masked fragments)
    v16h aq[4];
#pragma unroll
    for (int mt = 0; mt < 4; ++mt)
      aq[mt] = load_frag_a(qb, STRQ, mt * 16, 0, HC, lane);
    v8f S[4][4];
#pragma unroll
    for (int nt = 0; nt < 4; ++nt) {
      const v16h bk = load_frag_b(kb, STRQ, 1, nt * 16, 0, HC, lane);
#pragma unroll
      for (int mt = 0; mt < 4; ++mt)
        S[mt][nt] = __builtin_amdgcn_wmma_f32_16x16x32_f16(
            false, aq[mt], false, bk, (short)0, (v8f){}, false, false);
    }

    // row softmax entirely in registers (C/D layout: row=VGPR, col=lane%16)
    const int col15 = lane & 15;
#pragma unroll
    for (int mt = 0; mt < 4; ++mt) {
#pragma unroll
      for (int e = 0; e < 8; ++e) {
        float v4[4];
        float mx = -3.0e38f;
#pragma unroll
        for (int nt = 0; nt < 4; ++nt) {
          const int c = nt * 16 + col15;
          v4[nt] = (c < NTOK) ? S[mt][nt][e] : -3.0e38f;
          mx = fmaxf(mx, v4[nt]);
        }
        mx = wave_half_max(mx);
        float sum = 0.0f;
#pragma unroll
        for (int nt = 0; nt < 4; ++nt) {
          const int c = nt * 16 + col15;
          v4[nt] = (c < NTOK) ? __expf(v4[nt] - mx) : 0.0f;
          sum += v4[nt];
        }
        sum = wave_half_sum(sum);
        const float inv = 1.0f / sum;
#pragma unroll
        for (int nt = 0; nt < 4; ++nt) S[mt][nt][e] = v4[nt] * inv;
      }
    }

    // spill P to wave-private LDS scratch as f16 (C/D -> row-major)
#pragma unroll
    for (int mt = 0; mt < 4; ++mt)
#pragma unroll
      for (int nt = 0; nt < 4; ++nt)
#pragma unroll
        for (int e = 0; e < 8; ++e) {
          const int m = mt * 16 + e + ((lane >> 4) << 3);
          const int c = nt * 16 + col15;
          myP[m * STRP + c] = (_Float16)S[mt][nt][e];
        }
    asm volatile("s_wait_dscnt 0x0" ::: "memory");

    // O = P @ V   (M=64, N=16 (d<8 valid), K=64 in two 32-steps)
    v8f O[4];
#pragma unroll
    for (int mt = 0; mt < 4; ++mt) O[mt] = (v8f){};
#pragma unroll
    for (int ks2 = 0; ks2 < 2; ++ks2) {
      const v16h bv = load_frag_b(vb, 1, STRQ, 0, ks2 * 32, 64, lane);
#pragma unroll
      for (int mt = 0; mt < 4; ++mt) {
        const v16h ap = load_frag_a(myP, STRP, mt * 16, ks2 * 32, 64, lane);
        O[mt] = __builtin_amdgcn_wmma_f32_16x16x32_f16(
            false, ap, false, bv, (short)0, O[mt], false, false);
      }
    }

    // fused window-reverse store: (b, hpi*7+wh, wpi*7+ww, head*8+d)
    const int d = col15;
    if (d < HC) {
#pragma unroll
      for (int mt = 0; mt < 4; ++mt)
#pragma unroll
        for (int e = 0; e < 8; ++e) {
          const int m = mt * 16 + e + ((lane >> 4) << 3);
          if (m < NTOK) {
            const int wh = m / 7, ww = m % 7;
            const long long o =
                ((long long)bb * (HHW * HHW) +
                 (long long)((hpi * 7 + wh) * HHW + (wpi * 7 + ww))) * CH +
                head * HC + d;
            out[o] = O[mt][e];
          }
        }
    }
  }
}

extern "C" void kernel_launch(void* const* d_in, const int* in_sizes, int n_in,
                              void* d_out, int out_size, void* d_ws, size_t ws_size,
                              hipStream_t stream) {
  const float* x    = (const float*)d_in[0];
  const float* wq   = (const float*)d_in[1];
  const float* bq   = (const float*)d_in[2];
  float* out        = (float*)d_out;
  _Float16* wsW     = (_Float16*)d_ws;          // 768*256*2 = 384 KB scratch
  (void)in_sizes; (void)n_in; (void)ws_size; (void)out_size;

  // pre-pass: fp32 W -> f16 copy so the TDM can byte-copy tiles into LDS
  convert_w_f16<<<dim3(N3 * CH / (256 * 8)), dim3(256), 0, stream>>>(wq, wsW);

  dim3 grid(2048);   // B(32) * windows(64)
  dim3 block(256);   // 8 wave32
  swin_window_attn_gfx1250<<<grid, block, 0, stream>>>(x, wsW, bq, out);
}
